// PhotonicBasisActivationLayer_77369540870895
// MI455X (gfx1250) — matrix operand order repaired and checked by reading, not compiled, not run on hardware
//
#include <hip/hip_runtime.h>
#include <hip/hip_bf16.h>
#include <stdint.h>

typedef _Float16 v16h __attribute__((ext_vector_type(16)));
typedef _Float16 v8h  __attribute__((ext_vector_type(8)));
typedef float    v8f  __attribute__((ext_vector_type(8)));
typedef int      v4i  __attribute__((vector_size(16)));   // matches builtin param type
typedef __attribute__((address_space(1))) v4i* global_v4i_ptr;
typedef __attribute__((address_space(3))) v4i* shared_v4i_ptr;

#define B_ROWS  4096
#define IN_DIM  512
#define OUT_DIM 512
#define M_DIM   16
#define K_DIM   (IN_DIM * M_DIM)   // 8192

// GEMM tiling
#define BM     128
#define BN     128
#define BK     32
#define PADK   40                  // padded LDS row stride (halves) -> conflict-free b128 reads
#define SPLITK 4
#define KHALVES_PER_SLICE (K_DIM / SPLITK)       // 2048
#define KT_PER (KHALVES_PER_SLICE / BK)          // 64 k-steps per z-slice
#define NSTAGE 3

// ---------------------------------------------------------------------------
// CDNA5 async global->LDS copy (16B per lane), tracked by ASYNCcnt
// ---------------------------------------------------------------------------
__device__ __forceinline__ void async_copy16(const _Float16* gp, _Float16* sp) {
#if __has_builtin(__builtin_amdgcn_global_load_async_to_lds_b128)
    _Float16* gnc = const_cast<_Float16*>(gp);
    __builtin_amdgcn_global_load_async_to_lds_b128(
        (global_v4i_ptr)gnc, (shared_v4i_ptr)sp, 0, 0);
#else
    uint32_t lds_off = (uint32_t)(uintptr_t)(__attribute__((address_space(3))) void*)sp;
    asm volatile("global_load_async_to_lds_b128 %0, %1, off"
                 :: "v"(lds_off), "v"(gp) : "memory");
#endif
}

__device__ __forceinline__ void wait_async_le4() {
#if __has_builtin(__builtin_amdgcn_s_wait_asynccnt)
    __builtin_amdgcn_s_wait_asynccnt(4);
#else
    asm volatile("s_wait_asynccnt 0x4" ::: "memory");
#endif
}
__device__ __forceinline__ void wait_async_le0() {
#if __has_builtin(__builtin_amdgcn_s_wait_asynccnt)
    __builtin_amdgcn_s_wait_asynccnt(0);
#else
    asm volatile("s_wait_asynccnt 0x0" ::: "memory");
#endif
}

// ---------------------------------------------------------------------------
// Phase 0: zero the output (split-K accumulates with atomics)
// ---------------------------------------------------------------------------
__global__ __launch_bounds__(256)
void zero_out_f32(float4* __restrict__ p) {
    p[blockIdx.x * 256 + threadIdx.x] = make_float4(0.f, 0.f, 0.f, 0.f);
}

// ---------------------------------------------------------------------------
// Phase 1: basis[b,i,m] -> f16, written K-contiguous as A[4096][8192]
// ---------------------------------------------------------------------------
__global__ __launch_bounds__(256)
void photonic_basis_f16(const float* __restrict__ x,
                        const float* __restrict__ bc,   // (16,8)
                        _Float16* __restrict__ A) {
    int idx = blockIdx.x * 256 + threadIdx.x;           // (b,i) flat
    float xv = x[idx];
    float x2 = xv * xv, x3 = x2 * xv, x4 = x2 * x2;
    v16h out;
#pragma unroll
    for (int m = 0; m < M_DIM; ++m) {
        float b1 = bc[m * 8 + 0], b2 = bc[m * 8 + 1];
        float b3 = bc[m * 8 + 2], b4 = bc[m * 8 + 3];
        float b5 = bc[m * 8 + 4], b6 = bc[m * 8 + 5];
        float b7 = bc[m * 8 + 6], b8 = bc[m * 8 + 7];
        float inner   = __expf(b3 * xv) - 1.0f;              // expm1, arg >= ~0.01
        float powered = __expf(b4 * __logf(inner));          // inner^b4, inner > 0
        float y = b1 * __logf(1.0f + b2 * __logf(1.0f + powered))
                + b5 * xv + b6 * x2 + b7 * x3 + b8 * x4;
        out[m] = (_Float16)y;
    }
    *reinterpret_cast<v16h*>(A + (size_t)idx * M_DIM) = out;
}

// ---------------------------------------------------------------------------
// Phase 2: coeffs (OUT,IN,M) f32 -> f16, already [512][8192] K-contiguous
// ---------------------------------------------------------------------------
__global__ __launch_bounds__(256)
void coeffs_to_f16(const float* __restrict__ c, _Float16* __restrict__ o) {
    int idx = blockIdx.x * 256 + threadIdx.x;
    size_t base = (size_t)idx * 8;
    float4 lo = *(const float4*)(c + base);
    float4 hi = *(const float4*)(c + base + 4);
    v8h v;
    v[0] = (_Float16)lo.x; v[1] = (_Float16)lo.y;
    v[2] = (_Float16)lo.z; v[3] = (_Float16)lo.w;
    v[4] = (_Float16)hi.x; v[5] = (_Float16)hi.y;
    v[6] = (_Float16)hi.z; v[7] = (_Float16)hi.w;
    *(v8h*)(o + base) = v;
}

// ---------------------------------------------------------------------------
// Phase 3: Y[4096x512] += A_slice * B_slice^T, split-K over blockIdx.z.
// 256 threads = 8 waves (2x4), wave tile 64x32 = 4x2 WMMA accumulators.
// 3-stage async (global->LDS) pipeline, one barrier per K-step.
// ---------------------------------------------------------------------------
__global__ __launch_bounds__(256)
void wmma_gemm_f16(const _Float16* __restrict__ A,
                   const _Float16* __restrict__ Bm,
                   float* __restrict__ C) {
    __shared__ __align__(16) _Float16 As[NSTAGE][BM * PADK];
    __shared__ __align__(16) _Float16 Bs[NSTAGE][BN * PADK];

    const int tid  = threadIdx.x;
    const int wave = tid >> 5;
    const int lane = tid & 31;
    const int wm   = wave >> 2;        // 0..1 : 64-row slab
    const int wn   = wave & 3;         // 0..3 : 32-col slab
    const int lr   = lane & 15;
    const int lh   = lane >> 4;        // K-half selector: k = 16*lh + s (this code's order)

    const size_t rowA0 = (size_t)blockIdx.x * BM;
    const size_t rowB0 = (size_t)blockIdx.y * BN;
    const size_t kbase = (size_t)blockIdx.z * KHALVES_PER_SLICE;

    // cooperative copy mapping: 512 chunks (128 rows x 4 x 16B) per tile
    const int qa0 = tid, qa1 = tid + 256;
    const int ra0 = qa0 >> 2, ca0 = qa0 & 3;
    const int ra1 = qa1 >> 2, ca1 = qa1 & 3;
    const _Float16* gA0 = A  + (rowA0 + ra0) * K_DIM + kbase + ca0 * 8;
    const _Float16* gA1 = A  + (rowA0 + ra1) * K_DIM + kbase + ca1 * 8;
    const _Float16* gB0 = Bm + (rowB0 + ra0) * K_DIM + kbase + ca0 * 8;
    const _Float16* gB1 = Bm + (rowB0 + ra1) * K_DIM + kbase + ca1 * 8;
    const int la0 = ra0 * PADK + ca0 * 8;
    const int la1 = ra1 * PADK + ca1 * 8;

    v8f zero = {0.f, 0.f, 0.f, 0.f, 0.f, 0.f, 0.f, 0.f};
    v8f acc[4][2];
#pragma unroll
    for (int i = 0; i < 4; ++i)
#pragma unroll
        for (int j = 0; j < 2; ++j) acc[i][j] = zero;

    auto issueStage = [&](int s) {     // 4 async b128 per thread, no VGPR staging
        async_copy16(gA0, &As[s][la0]); gA0 += BK;
        async_copy16(gA1, &As[s][la1]); gA1 += BK;
        async_copy16(gB0, &Bs[s][la0]); gB0 += BK;
        async_copy16(gB1, &Bs[s][la1]); gB1 += BK;
    };

    auto computeStage = [&](int s) {
        v16h bfr[2];
#pragma unroll
        for (int j = 0; j < 2; ++j) {
            // 16-bit B 32x16: lanes 0-15 -> K 0..15, lanes 16-31 -> K 16..31
            int col = wn * 32 + j * 16 + lr;
            const v8h* p = (const v8h*)(&Bs[s][col * PADK + lh * 16]);
            v8h lo = p[0], hi = p[1];
            bfr[j] = __builtin_shufflevector(lo, hi, 0,1,2,3,4,5,6,7,8,9,10,11,12,13,14,15);
        }
        v16h afr[4];
#pragma unroll
        for (int i = 0; i < 4; ++i) {
            // 16-bit A 16x32: lane = K base..base+15, base = lh*16
            int row = wm * 64 + i * 16 + lr;
            v8h lo = *(const v8h*)(&As[s][row * PADK + lh * 16]);
            v8h hi = *(const v8h*)(&As[s][row * PADK + lh * 16 + 8]);
            afr[i] = __builtin_shufflevector(lo, hi, 0,1,2,3,4,5,6,7,8,9,10,11,12,13,14,15);
        }
#pragma unroll
        for (int i = 0; i < 4; ++i)
#pragma unroll
            for (int j = 0; j < 2; ++j)
                acc[i][j] = __builtin_amdgcn_wmma_f32_16x16x32_f16(
                    false, afr[i], false, bfr[j], (short)0, acc[i][j], false, false);
    };

    // prologue: two stages in flight
    issueStage(0);
    issueStage(1);

    int s = 0;
#pragma unroll 1
    for (int kt = 0; kt < KT_PER - 1; ++kt) {
        wait_async_le4();              // stage s fully landed (in-order completion)
        __syncthreads();               // ...for every thread in the block
        computeStage(s);
        if (kt + 2 < KT_PER) issueStage((s + 2) % NSTAGE);  // overwrites stage read at kt-1
        s = (s + 1) % NSTAGE;
    }
    wait_async_le0();                  // last stage
    __syncthreads();
    computeStage(s);

    // epilogue: split-K accumulate via global_atomic_add_f32
    // C layout per ISA: lane l -> col = l%16, rows lh*8 + r in v[r]
#pragma unroll
    for (int i = 0; i < 4; ++i)
#pragma unroll
        for (int j = 0; j < 2; ++j) {
            size_t r0 = rowA0 + wm * 64 + i * 16 + lh * 8;
            int    c0 = (int)rowB0 + wn * 32 + j * 16 + lr;
#pragma unroll
            for (int r = 0; r < 8; ++r)
                unsafeAtomicAdd(&C[(r0 + r) * OUT_DIM + c0], acc[i][j][r]);
        }
}

// ---------------------------------------------------------------------------
extern "C" void kernel_launch(void* const* d_in, const int* in_sizes, int n_in,
                              void* d_out, int out_size, void* d_ws, size_t ws_size,
                              hipStream_t stream) {
    const float* x      = (const float*)d_in[0];   // (4096, 512)
    const float* coeffs = (const float*)d_in[1];   // (512, 512, 16) == [512][8192]
    const float* bcoef  = (const float*)d_in[2];   // (16, 8)
    float* out = (float*)d_out;                    // (4096, 512) f32

    // workspace: A f16 (64 MiB) then Bf f16 (8 MiB)
    _Float16* Af = (_Float16*)d_ws;
    _Float16* Bf = (_Float16*)((char*)d_ws + (size_t)B_ROWS * K_DIM * sizeof(_Float16));

    zero_out_f32<<<(B_ROWS * OUT_DIM / 4) / 256, 256, 0, stream>>>((float4*)out);
    photonic_basis_f16<<<(B_ROWS * IN_DIM) / 256, 256, 0, stream>>>(x, bcoef, Af);
    coeffs_to_f16<<<(OUT_DIM * K_DIM / 8) / 256, 256, 0, stream>>>(coeffs, Bf);

    dim3 grid(B_ROWS / BM, OUT_DIM / BN, SPLITK);  // 32 x 4 x 4 = 512 workgroups
    wmma_gemm_f16<<<grid, 256, 0, stream>>>(Af, Bf, out);
}
